// KANConvTranspose2d_59296318488612
// MI455X (gfx1250) — matrix operand order repaired and checked by reading, not compiled, not run on hardware
//
#include <hip/hip_runtime.h>
#include <stdint.h>

// ---------------- problem constants ----------------
static constexpr int CIN   = 16;
static constexpr int IN_F  = 2304;              // CIN*9*16
static constexpr int SPL   = 8;                 // GRID_SIZE + SPLINE_ORDER
static constexpr int KDIM  = IN_F * (1 + SPL);  // 20736 (silu cols + spline cols)
static constexpr int OUT_F = 4608;
static constexpr int BATCH = 64;
static constexpr int NKSTEP = KDIM / 32;        // 648 k-steps of 32
static constexpr int BASE_KSTEPS = IN_F / 32;   // 72 (base-weight region)

// workspace layout (bytes); A buffers padded 4KB for 1-step load lookahead
static constexpr size_t ABYTES = (size_t)BATCH * KDIM * 2;      // 2,654,208
static constexpr size_t APAD   = 4096;
static constexpr size_t OFF_ALO = ABYTES + APAD;
static constexpr size_t OFF_Y   = 2 * (ABYTES + APAD);

typedef __attribute__((ext_vector_type(16))) __bf16 v16bf;
typedef __attribute__((ext_vector_type(8)))  float  v8f;

__device__ __forceinline__ uint16_t f2bf(float f) {
  __bf16 b = (__bf16)f;                       // RNE hardware convert
  return __builtin_bit_cast(uint16_t, b);
}
__device__ __forceinline__ float bf2f(uint16_t u) {
  return (float)__builtin_bit_cast(__bf16, u);
}

// =====================================================================
// Kernel 1: unfold + silu + cubic B-spline bases -> bf16 hi/lo activations
// A layout: row-major [batch m][k], k<2304 = silu(u), k>=2304 = bases i*8+t
// =====================================================================
__global__ void prep_act(const float* __restrict__ x, const float* __restrict__ grid,
                         uint16_t* __restrict__ Ahi, uint16_t* __restrict__ Alo) {
  int t = blockIdx.x * blockDim.x + threadIdx.x;
  if (t >= BATCH * IN_F) return;
  int b = t / IN_F;
  int i = t - b * IN_F;                 // i = c*144 + (ki*3+kj)*16 + (oh*4+ow)
  int c  = i / 144;
  int kk = (i >> 4) % 9;
  int s  = i & 15;
  int ki = kk / 3, kj = kk % 3;
  int oh = s >> 2, ow = s & 3;
  int row = oh * 2 + ki - 1;            // stride 2, pad 1
  int col = ow * 2 + kj - 1;
  float u = 0.f;
  if (row >= 0 && row < 8 && col >= 0 && col < 8)
    u = x[((b * CIN + c) * 8 + row) * 8 + col];

  float sil = u / (1.f + __expf(-u));   // silu

  // degree-3 Cox–de Boor on 12 knots -> 8 bases (matches reference exactly)
  float g[12];
  #pragma unroll
  for (int q = 0; q < 12; ++q) g[q] = grid[i * 12 + q];
  float bas[11];
  #pragma unroll
  for (int q = 0; q < 11; ++q) bas[q] = (u >= g[q] && u < g[q + 1]) ? 1.f : 0.f;
  #pragma unroll
  for (int k = 1; k <= 3; ++k) {
    #pragma unroll
    for (int q = 0; q < 11 - 3; ++q) {  // unrolled upper bound; guard inside
      if (q < 11 - k) {
        bas[q] = (u - g[q]) / (g[q + k] - g[q]) * bas[q]
               + (g[q + k + 1] - u) / (g[q + k + 1] - g[q + 1]) * bas[q + 1];
      }
    }
  }

  size_t ro = (size_t)b * KDIM;
  {
    uint16_t h = f2bf(sil);
    Ahi[ro + i] = h;
    Alo[ro + i] = f2bf(sil - bf2f(h));
  }
  size_t sp = ro + IN_F + (size_t)i * 8;
  #pragma unroll
  for (int q = 0; q < 8; ++q) {
    uint16_t h = f2bf(bas[q]);
    Ahi[sp + q] = h;
    Alo[sp + q] = f2bf(bas[q] - bf2f(h));
  }
}

// =====================================================================
// GEMM helpers
// A fragment k order used here (contiguous, same for B): lanes 0-15 hold K 0-15,
// lanes 16-31 hold K 16-31 -> two b128 loads at aoff, aoff+8.
// =====================================================================
struct AFrag {
  union { uint4 q[2]; v16bf v; } h[4], l[4];   // hi/lo for 4 m-tiles
};

__device__ __forceinline__ void load_af(AFrag& f,
                                        const uint16_t* __restrict__ Ahi,
                                        const uint16_t* __restrict__ Alo,
                                        size_t aoff) {
  #pragma unroll
  for (int mt = 0; mt < 4; ++mt) {
    const size_t off = aoff + (size_t)(mt * 16) * KDIM;
    f.h[mt].q[0] = *(const uint4*)(Ahi + off);
    f.h[mt].q[1] = *(const uint4*)(Ahi + off + 8);
    f.l[mt].q[0] = *(const uint4*)(Alo + off);
    f.l[mt].q[1] = *(const uint4*)(Alo + off + 8);
  }
}

__device__ __forceinline__ void conv_b(const float* wf, v16bf& bh, v16bf& bl) {
  union { uint16_t u[16]; v16bf v; } H, L;
  #pragma unroll
  for (int e = 0; e < 16; ++e) {
    uint16_t h = f2bf(wf[e]);
    H.u[e] = h;
    L.u[e] = f2bf(wf[e] - bf2f(h));
  }
  bh = H.v; bl = L.v;
}

// 12 compensated WMMAs: hi*hi, hi*lo, lo*hi (lo*lo ~2^-16, dropped)
__device__ __forceinline__ void do_wmma(const AFrag& f, const v16bf& bh,
                                        const v16bf& bl, v8f acc[4]) {
  #pragma unroll
  for (int mt = 0; mt < 4; ++mt)
    acc[mt] = __builtin_amdgcn_wmma_f32_16x16x32_bf16(
        false, f.h[mt].v, false, bh, (short)0, acc[mt], false, false);
  #pragma unroll
  for (int mt = 0; mt < 4; ++mt)
    acc[mt] = __builtin_amdgcn_wmma_f32_16x16x32_bf16(
        false, f.h[mt].v, false, bl, (short)0, acc[mt], false, false);
  #pragma unroll
  for (int mt = 0; mt < 4; ++mt)
    acc[mt] = __builtin_amdgcn_wmma_f32_16x16x32_bf16(
        false, f.l[mt].v, false, bh, (short)0, acc[mt], false, false);
}

// =====================================================================
// Kernel 2: fused dual-GEMM y = A(64x20736) * W^T(20736x4608)
// One workgroup (4 waves) per 16-column N strip; waves split K 4-way
// (stride-4 interleave, scalar wave id). Weights streamed from HBM exactly
// once. A fragments are double-buffered across k-steps (fa/fb) so the next
// step's 16 b128 loads are in flight while the current 12 WMMAs execute —
// both buffers live => no register-recycled serialized load chains.
// =====================================================================
__global__ void __launch_bounds__(128, 1)
kan_gemm(const uint16_t* __restrict__ Ahi, const uint16_t* __restrict__ Alo,
         const float* __restrict__ Wbase, const float* __restrict__ Wspl,
         const float* __restrict__ Wscl, float* __restrict__ y) {
  const int ntile = blockIdx.x;         // 0..287
  const int tid   = threadIdx.x;
  const int wid   = __builtin_amdgcn_readfirstlane(tid >> 5);  // 0..3, scalar
  const int lane  = tid & 31;
  const int lan16 = lane & 15;
  const int khalf = lane >> 4;
  const int n     = ntile * 16 + lan16; // output column owned by this lane

  const float* wp = Wbase + (size_t)n * IN_F + wid * 32 + khalf * 16;
  const float* sp = Wspl + (size_t)n * (IN_F * SPL) + wid * 32 + khalf * 16;
  const float* sc = Wscl + (size_t)n * IN_F;

  // per-lane A offset for ks=wid; each k-step advances 4*32 elements
  size_t aoff = (size_t)lan16 * KDIM + (size_t)khalf * 16 + (size_t)wid * 32;
  constexpr size_t ASTEP = 4 * 32;

  v8f acc[4] = {v8f{}, v8f{}, v8f{}, v8f{}};   // 4 m-tiles (M = 64)

  AFrag fa, fb;
  load_af(fa, Ahi, Alo, aoff);                 // prologue: first k-step

  // ---------------- phase 1: base weights, 9 double-steps (ks wid..wid+68) ----
  for (int it = 0; it < 9; ++it) {
    union { float4 q[4]; float f[16]; } w0, w1;
    {
      const float4* p = (const float4*)wp;
      w0.q[0] = p[0]; w0.q[1] = p[1]; w0.q[2] = p[2]; w0.q[3] = p[3];
    }
    __builtin_prefetch(wp + 256, 0, 1);
    load_af(fb, Ahi, Alo, aoff + ASTEP);
    v16bf bh, bl;
    conv_b(w0.f, bh, bl);
    do_wmma(fa, bh, bl, acc);

    {
      const float4* p = (const float4*)(wp + 128);
      w1.q[0] = p[0]; w1.q[1] = p[1]; w1.q[2] = p[2]; w1.q[3] = p[3];
    }
    __builtin_prefetch(wp + 384, 0, 1);
    load_af(fa, Ahi, Alo, aoff + 2 * ASTEP);
    conv_b(w1.f, bh, bl);
    do_wmma(fb, bh, bl, acc);

    wp += 256;                 // 2 k-steps * (4-wave stride * 32)
    aoff += 2 * ASTEP;
  }

  // ---------------- phase 2: spline weights, 72 double-steps ----------------
  // scaler: 16 consecutive k cover 2 input features; i0 = (ks-72)*4 + khalf*2
  int i0 = wid * 4 + khalf * 2;
  for (int it = 0; it < 72; ++it) {
    union { float4 q[4]; float f[16]; } w0, w1;
    {
      const float4* p = (const float4*)sp;
      w0.q[0] = p[0]; w0.q[1] = p[1]; w0.q[2] = p[2]; w0.q[3] = p[3];
    }
    const float2 s01 = *(const float2*)(sc + i0);
    __builtin_prefetch(sp + 256, 0, 1);
    load_af(fb, Ahi, Alo, aoff + ASTEP);
    #pragma unroll
    for (int e = 0; e < 8; ++e)  w0.f[e] *= s01.x;
    #pragma unroll
    for (int e = 8; e < 16; ++e) w0.f[e] *= s01.y;
    v16bf bh, bl;
    conv_b(w0.f, bh, bl);
    do_wmma(fa, bh, bl, acc);

    {
      const float4* p = (const float4*)(sp + 128);
      w1.q[0] = p[0]; w1.q[1] = p[1]; w1.q[2] = p[2]; w1.q[3] = p[3];
    }
    const float2 s23 = *(const float2*)(sc + i0 + 16);
    __builtin_prefetch(sp + 384, 0, 1);
    load_af(fa, Ahi, Alo, aoff + 2 * ASTEP);   // last iter: reads into 4KB pad
    #pragma unroll
    for (int e = 0; e < 8; ++e)  w1.f[e] *= s23.x;
    #pragma unroll
    for (int e = 8; e < 16; ++e) w1.f[e] *= s23.y;
    conv_b(w1.f, bh, bl);
    do_wmma(fb, bh, bl, acc);

    sp += 256;
    i0 += 32;
    aoff += 2 * ASTEP;
  }

  // ---- deterministic cross-wave K reduction via LDS (fixed order) ----
  __shared__ float red[4][4][256];      // [wave][mtile][r*32+lane] = 16KB
  #pragma unroll
  for (int mt = 0; mt < 4; ++mt)
    #pragma unroll
    for (int r = 0; r < 8; ++r)
      red[wid][mt][r * 32 + lane] = acc[mt][r];
  __syncthreads();

  const int mt = wid;                   // wave w finalizes m-tile w
  #pragma unroll
  for (int r = 0; r < 8; ++r) {
    float v = red[0][mt][r * 32 + lane] + red[1][mt][r * 32 + lane]
            + red[2][mt][r * 32 + lane] + red[3][mt][r * 32 + lane];
    // C/D layout: lanes 0-15 -> M=r, lanes 16-31 -> M=r+8; N = lane&15
    int m = mt * 16 + r + (khalf << 3);
    y[(size_t)m * OUT_F + ntile * 16 + lan16] = v;
  }
}

// =====================================================================
// Kernel 3: fold (gather form, deterministic): out(64,8,16,16)
// =====================================================================
__global__ void fold_out(const float* __restrict__ y, float* __restrict__ out, int total) {
  int idx = blockIdx.x * blockDim.x + threadIdx.x;
  if (idx >= total) return;
  int w  = idx & 15;
  int h  = (idx >> 4) & 15;
  int co = (idx >> 8) & 7;
  int b  = idx >> 11;
  float s = 0.f;
  #pragma unroll
  for (int ki = 0; ki < 3; ++ki) {
    int hh = h + 1 - ki;                 // padded row - stride mapping (S=2,P=1)
    if (hh < 0 || (hh & 1)) continue;
    int oh = hh >> 1;
    if (oh >= 8) continue;
    #pragma unroll
    for (int kj = 0; kj < 3; ++kj) {
      int ww = w + 1 - kj;
      if (ww < 0 || (ww & 1)) continue;
      int ow = ww >> 1;
      if (ow >= 8) continue;
      int ncol = ((co * 9 + ki * 3 + kj) << 6) + (oh << 3) + ow;
      s += y[(size_t)b * OUT_F + ncol];
    }
  }
  out[idx] = s;
}

// =====================================================================
// Launch: prep -> gemm -> fold, all on `stream` (graph-capture safe).
// Workspace: Ahi (2.65MB+pad) | Alo (2.65MB+pad) | y (1.18MB) ~= 6.5 MB
// =====================================================================
extern "C" void kernel_launch(void* const* d_in, const int* in_sizes, int n_in,
                              void* d_out, int out_size, void* d_ws, size_t ws_size,
                              hipStream_t stream) {
  const float* x    = (const float*)d_in[0];
  const float* wbse = (const float*)d_in[1];
  const float* wspl = (const float*)d_in[2];
  const float* wscl = (const float*)d_in[3];
  const float* grid = (const float*)d_in[4];

  char* ws = (char*)d_ws;
  uint16_t* Ahi = (uint16_t*)ws;
  uint16_t* Alo = (uint16_t*)(ws + OFF_ALO);
  float*    yb  = (float*)(ws + OFF_Y);
  float*    out = (float*)d_out;

  prep_act<<<(BATCH * IN_F) / 256, 256, 0, stream>>>(x, grid, Ahi, Alo);
  kan_gemm<<<OUT_F / 16, 128, 0, stream>>>(Ahi, Alo, wbse, wspl, wscl, yb);
  fold_out<<<(out_size + 255) / 256, 256, 0, stream>>>(yb, out, out_size);
}
